// Core_20796231647369
// MI455X (gfx1250) — matrix operand order repaired and checked by reading, not compiled, not run on hardware
//
#include <hip/hip_runtime.h>
#include <hip/hip_bf16.h>

#define T_DIM 8192
#define O_DIM 1024
#define H_DIM 2048
#define A_DIM 512
#define G3H   (3 * H_DIM)

#define SCAN_BLOCKS 128
#define SCAN_THREADS 512        // 16 waves; j = blockIdx*16 + wave
#define KCHUNK 64               // h-slice per lane (2048 / 32 lanes)

typedef __attribute__((ext_vector_type(16))) __bf16 v16bf;
typedef __attribute__((ext_vector_type(8)))  __bf16 v8bf;
typedef __attribute__((ext_vector_type(4)))  __bf16 v4bf;
typedef __attribute__((ext_vector_type(8)))  float  v8f;

// ---------------- elementwise fp32 -> bf16 hi/lo plane split ----------------
__global__ __launch_bounds__(256) void split_planes(
    const float* __restrict__ src, __bf16* __restrict__ hi,
    __bf16* __restrict__ lo, long long n4) {
  long long i = (long long)blockIdx.x * 256 + threadIdx.x;
  long long stride = (long long)gridDim.x * 256;
  for (; i < n4; i += stride) {
    float4 x = *(const float4*)(src + 4 * i);
    v4bf h, l;
    h[0] = (__bf16)x.x; l[0] = (__bf16)(x.x - (float)h[0]);
    h[1] = (__bf16)x.y; l[1] = (__bf16)(x.y - (float)h[1]);
    h[2] = (__bf16)x.z; l[2] = (__bf16)(x.z - (float)h[2]);
    h[3] = (__bf16)x.w; l[3] = (__bf16)(x.w - (float)h[3]);
    *(v4bf*)(hi + 4 * i) = h;
    *(v4bf*)(lo + 4 * i) = l;
  }
}

// C[M,N] = A[M,K] @ W[N,K]^T + bias[N], bf16 hi/lo planes, 3 WMMA per tile-k.
// One wave: 16(M) x 64(N) tile. M%16==0, N%64==0, K%32==0.
__global__ __launch_bounds__(256) void wmma_gemm_planes(
    const __bf16* __restrict__ Ahi, const __bf16* __restrict__ Alo,
    const __bf16* __restrict__ Whi, const __bf16* __restrict__ Wlo,
    const float* __restrict__ bias, float* __restrict__ Cmat,
    int M, int N, int K) {
  const int lane  = threadIdx.x & 31;
  const int wave  = threadIdx.x >> 5;
  const int gwave = blockIdx.x * 8 + wave;
  const int ntiles = N >> 6;
  const int mt = gwave / ntiles;
  const int nt = gwave % ntiles;
  if (mt >= (M >> 4)) return;

  const int l15 = lane & 15;   // M row (A) / N col (B,C)
  const int khi = lane >> 4;   // K-half selector (A,B) / M-half (C,D)

  v8f acc[4];
#pragma unroll
  for (int s = 0; s < 4; ++s) acc[s] = v8f{};

  const long long abase = (long long)(mt * 16 + l15) * K;

  for (int k0 = 0; k0 < K; k0 += 32) {
    const long long ka = abase + k0 + khi * 8;
    v8bf ah0 = *(const v8bf*)(Ahi + ka);
    v8bf ah1 = *(const v8bf*)(Ahi + ka + 16);
    v8bf al0 = *(const v8bf*)(Alo + ka);
    v8bf al1 = *(const v8bf*)(Alo + ka + 16);
    v16bf a_hi = __builtin_shufflevector(ah0, ah1, 0, 1, 2, 3, 4, 5, 6, 7,
                                         8, 9, 10, 11, 12, 13, 14, 15);
    v16bf a_lo = __builtin_shufflevector(al0, al1, 0, 1, 2, 3, 4, 5, 6, 7,
                                         8, 9, 10, 11, 12, 13, 14, 15);
#pragma unroll
    for (int s = 0; s < 4; ++s) {
      const long long kb =
          (long long)(nt * 64 + s * 16 + l15) * K + k0 + khi * 8;
      v16bf b_hi = __builtin_shufflevector(*(const v8bf*)(Whi + kb),
                                           *(const v8bf*)(Whi + kb + 16),
                                           0, 1, 2, 3, 4, 5, 6, 7,
                                           8, 9, 10, 11, 12, 13, 14, 15);
      v16bf b_lo = __builtin_shufflevector(*(const v8bf*)(Wlo + kb),
                                           *(const v8bf*)(Wlo + kb + 16),
                                           0, 1, 2, 3, 4, 5, 6, 7,
                                           8, 9, 10, 11, 12, 13, 14, 15);
      acc[s] = __builtin_amdgcn_wmma_f32_16x16x32_bf16(
          false, a_hi, false, b_hi, (short)0, acc[s], false, false);
      acc[s] = __builtin_amdgcn_wmma_f32_16x16x32_bf16(
          false, a_lo, false, b_hi, (short)0, acc[s], false, false);
      acc[s] = __builtin_amdgcn_wmma_f32_16x16x32_bf16(
          false, a_hi, false, b_lo, (short)0, acc[s], false, false);
    }
  }

#pragma unroll
  for (int s = 0; s < 4; ++s) {
    const int n = nt * 64 + s * 16 + l15;
    const float b = bias[n];
#pragma unroll
    for (int r = 0; r < 8; ++r) {
      const int m = mt * 16 + r + 8 * khi;
      Cmat[(long long)m * N + n] = acc[s][r] + b;
    }
  }
}

// ---------------- persistent GRU scan, register-resident W_hh ----------------
__device__ __forceinline__ void grid_barrier(unsigned* __restrict__ count,
                                             unsigned* __restrict__ gen,
                                             unsigned nblocks) {
  __threadfence();
  __syncthreads();
  if (threadIdx.x == 0) {
    unsigned g = __hip_atomic_load(gen, __ATOMIC_RELAXED, __HIP_MEMORY_SCOPE_AGENT);
    unsigned arrived = __hip_atomic_fetch_add(count, 1u, __ATOMIC_ACQ_REL,
                                              __HIP_MEMORY_SCOPE_AGENT);
    if (arrived == nblocks - 1) {
      __hip_atomic_store(count, 0u, __ATOMIC_RELAXED, __HIP_MEMORY_SCOPE_AGENT);
      __hip_atomic_fetch_add(gen, 1u, __ATOMIC_RELEASE, __HIP_MEMORY_SCOPE_AGENT);
    } else {
      while (__hip_atomic_load(gen, __ATOMIC_ACQUIRE, __HIP_MEMORY_SCOPE_AGENT) == g) {
        __builtin_amdgcn_s_sleep(2);
      }
    }
  }
  __syncthreads();
}

__device__ __forceinline__ float sigmoidf_(float x) {
  return 1.f / (1.f + expf(-x));
}

__global__ __launch_bounds__(SCAN_THREADS) void gru_scan(
    const float* __restrict__ W_hh,   // [3H, H]
    const float* __restrict__ b_hh,   // [3H]
    const float* __restrict__ xi,     // [T, 3H]
    float* __restrict__ hbuf,         // [2, H] (zeroed)
    float* __restrict__ hs,           // [T, H]
    unsigned* __restrict__ bar) {     // {count, gen} (zeroed)
  __shared__ float h_s[H_DIM];
  const int tid   = threadIdx.x;
  const int wave  = tid >> 5;
  const int lane  = tid & 31;
  const int j     = blockIdx.x * (SCAN_THREADS / 32) + wave;  // hidden channel
  const int kbase = lane * KCHUNK;

  // One-time preload: this lane's 3 x 64 fp32 weight slice into VGPRs.
  float w0[KCHUNK], w1[KCHUNK], w2[KCHUNK];
  {
    const float* r0 = W_hh + (long long)j * H_DIM + kbase;
    const float* r1 = W_hh + (long long)(H_DIM + j) * H_DIM + kbase;
    const float* r2 = W_hh + (long long)(2 * H_DIM + j) * H_DIM + kbase;
#pragma unroll
    for (int k = 0; k < KCHUNK; k += 4) {
      *(float4*)&w0[k] = *(const float4*)(r0 + k);
      *(float4*)&w1[k] = *(const float4*)(r1 + k);
      *(float4*)&w2[k] = *(const float4*)(r2 + k);
    }
  }
  const float bh0 = b_hh[j];
  const float bh1 = b_hh[H_DIM + j];
  const float bh2 = b_hh[2 * H_DIM + j];

  // LDS byte address of this thread's h_s chunk (generic AS3 pointer: low 32
  // bits are the LDS offset, per ISA 10.2 aperture rules).
  const unsigned lds_dst =
      (unsigned)(unsigned long long)(const void*)&h_s[tid * 4];

  for (int t = 0; t < T_DIM; ++t) {
    const float* h_cur = hbuf + (long long)(t & 1) * H_DIM;
    float*       h_nxt = hbuf + (long long)((t + 1) & 1) * H_DIM;

    // Broadcast h into LDS: one async 16B global->LDS transfer per thread
    // (ASYNCcnt-tracked, no VGPR staging).
    {
      const float* gsrc = h_cur + tid * 4;
      asm volatile("global_load_async_to_lds_b128 %0, %1, off"
                   :
                   : "v"(lds_dst), "v"(gsrc)
                   : "memory");
    }

    if (lane == 0 && t + 1 < T_DIM) {   // overlap with the async copy
      const float* nx = xi + (long long)(t + 1) * G3H;
      __builtin_prefetch(nx + j, 0, 1);
      __builtin_prefetch(nx + H_DIM + j, 0, 1);
      __builtin_prefetch(nx + 2 * H_DIM + j, 0, 1);
    }

    asm volatile("s_wait_asynccnt 0x0" ::: "memory");
    __syncthreads();

    float s0 = 0.f, s1 = 0.f, s2 = 0.f;
#pragma unroll
    for (int k = 0; k < KCHUNK; k += 4) {
      float4 hv = *(const float4*)(h_s + kbase + k);
      s0 = fmaf(w0[k], hv.x, s0); s0 = fmaf(w0[k + 1], hv.y, s0);
      s0 = fmaf(w0[k + 2], hv.z, s0); s0 = fmaf(w0[k + 3], hv.w, s0);
      s1 = fmaf(w1[k], hv.x, s1); s1 = fmaf(w1[k + 1], hv.y, s1);
      s1 = fmaf(w1[k + 2], hv.z, s1); s1 = fmaf(w1[k + 3], hv.w, s1);
      s2 = fmaf(w2[k], hv.x, s2); s2 = fmaf(w2[k + 1], hv.y, s2);
      s2 = fmaf(w2[k + 2], hv.z, s2); s2 = fmaf(w2[k + 3], hv.w, s2);
    }
#pragma unroll
    for (int m = 1; m < 32; m <<= 1) {
      s0 += __shfl_xor(s0, m);
      s1 += __shfl_xor(s1, m);
      s2 += __shfl_xor(s2, m);
    }

    if (lane == 0) {
      const float* xi_t = xi + (long long)t * G3H;
      float r = sigmoidf_(xi_t[j] + s0 + bh0);
      float z = sigmoidf_(xi_t[H_DIM + j] + s1 + bh1);
      float n = tanhf(xi_t[2 * H_DIM + j] + r * (s2 + bh2));
      float hn = (1.f - z) * n + z * h_s[j];
      h_nxt[j] = hn;
      hs[(long long)t * H_DIM + j] = hn;
    }

    grid_barrier(&bar[0], &bar[1], SCAN_BLOCKS);
  }
}

__global__ void init_state(float* hbuf, unsigned* bar) {
  int i = threadIdx.x;
  for (int k = i; k < 2 * H_DIM; k += 256) hbuf[k] = 0.f;
  if (i < 2) bar[i] = 0u;
}

// ---------------- host orchestration ----------------
extern "C" void kernel_launch(void* const* d_in, const int* in_sizes, int n_in,
                              void* d_out, int out_size, void* d_ws, size_t ws_size,
                              hipStream_t stream) {
  (void)in_sizes; (void)n_in; (void)out_size; (void)ws_size;
  const float* obs  = (const float*)d_in[0];
  const float* W_ih = (const float*)d_in[1];
  const float* W_hh = (const float*)d_in[2];
  const float* b_ih = (const float*)d_in[3];
  const float* b_hh = (const float*)d_in[4];
  const float* W_o  = (const float*)d_in[5];
  const float* b_o  = (const float*)d_in[6];
  const float* W_d  = (const float*)d_in[7];
  const float* b_d  = (const float*)d_in[8];
  float* out = (float*)d_out;

  char* ws = (char*)d_ws;
  size_t cur = 0;
  auto take = [&](size_t bytes) {
    void* p = ws + cur;
    cur = (cur + bytes + 255) & ~(size_t)255;
    return p;
  };
  float*    xi     = (float*)take((size_t)T_DIM * G3H * 4);
  float*    hs     = (float*)take((size_t)T_DIM * H_DIM * 4);
  float*    hbuf   = (float*)take((size_t)2 * H_DIM * 4);
  unsigned* bar    = (unsigned*)take(256);
  __bf16*   obs_hi = (__bf16*)take((size_t)T_DIM * O_DIM * 2);
  __bf16*   obs_lo = (__bf16*)take((size_t)T_DIM * O_DIM * 2);
  __bf16*   wih_hi = (__bf16*)take((size_t)G3H * O_DIM * 2);
  __bf16*   wih_lo = (__bf16*)take((size_t)G3H * O_DIM * 2);
  __bf16*   wo_hi  = (__bf16*)take((size_t)A_DIM * H_DIM * 2);
  __bf16*   wo_lo  = (__bf16*)take((size_t)A_DIM * H_DIM * 2);
  __bf16*   wd_hi  = (__bf16*)take((size_t)A_DIM * H_DIM * 2);
  __bf16*   wd_lo  = (__bf16*)take((size_t)A_DIM * H_DIM * 2);
  __bf16*   hs_hi  = (__bf16*)take((size_t)T_DIM * H_DIM * 2);
  __bf16*   hs_lo  = (__bf16*)take((size_t)T_DIM * H_DIM * 2);

  init_state<<<1, 256, 0, stream>>>(hbuf, bar);

  // prep: fp32 -> bf16 hi/lo planes (memory-bound, one pass each)
  split_planes<<<4096, 256, 0, stream>>>(obs,  obs_hi, obs_lo,
                                         (long long)T_DIM * O_DIM / 4);
  split_planes<<<4096, 256, 0, stream>>>(W_ih, wih_hi, wih_lo,
                                         (long long)G3H * O_DIM / 4);
  split_planes<<<1024, 256, 0, stream>>>(W_o,  wo_hi,  wo_lo,
                                         (long long)A_DIM * H_DIM / 4);
  split_planes<<<1024, 256, 0, stream>>>(W_d,  wd_hi,  wd_lo,
                                         (long long)A_DIM * H_DIM / 4);

  // xi = obs @ W_ih^T + b_ih : M=8192, N=6144, K=1024
  {
    int waves = (T_DIM / 16) * (G3H / 64);
    wmma_gemm_planes<<<waves / 8, 256, 0, stream>>>(
        obs_hi, obs_lo, wih_hi, wih_lo, b_ih, xi, T_DIM, G3H, O_DIM);
  }

  gru_scan<<<SCAN_BLOCKS, SCAN_THREADS, 0, stream>>>(W_hh, b_hh, xi, hbuf, hs, bar);

  split_planes<<<8192, 256, 0, stream>>>(hs, hs_hi, hs_lo,
                                         (long long)T_DIM * H_DIM / 4);

  // o_logits / d_logits : M=8192, N=512, K=2048
  {
    int waves = (T_DIM / 16) * (A_DIM / 64);
    wmma_gemm_planes<<<waves / 8, 256, 0, stream>>>(
        hs_hi, hs_lo, wo_hi, wo_lo, b_o, out, T_DIM, A_DIM, H_DIM);
    wmma_gemm_planes<<<waves / 8, 256, 0, stream>>>(
        hs_hi, hs_lo, wd_hi, wd_lo, b_d, out + (size_t)T_DIM * A_DIM,
        T_DIM, A_DIM, H_DIM);
  }
}
